// VoxelResBackBone8x_71184787964530
// MI455X (gfx1250) — matrix operand order repaired and checked by reading, not compiled, not run on hardware
//
#include <hip/hip_runtime.h>

// CDNA5 / gfx1250, wave32. WMMA f16 -> f32 accumulate.
typedef __attribute__((ext_vector_type(16))) _Float16 v16h;
typedef __attribute__((ext_vector_type(8)))  float    v8f;

// ---------------------------------------------------------------------------
// utility kernels
// ---------------------------------------------------------------------------
__global__ void k_fill_i32(int* __restrict__ p, int v, int n) {
  int i = blockIdx.x * blockDim.x + threadIdx.x;
  if (i < n) p[i] = v;
}

__global__ void k_fill_f32(float* __restrict__ p, float v, int n) {
  int i = blockIdx.x * blockDim.x + threadIdx.x;
  if (i < n) p[i] = v;
}

// Fold eval-mode BN (eps=1e-3) + optional conv bias into per-channel scale/shift.
__global__ void k_fold_bn(const float* __restrict__ g, const float* __restrict__ b,
                          const float* __restrict__ m, const float* __restrict__ v,
                          const float* __restrict__ bias,
                          float* __restrict__ scale, float* __restrict__ shift, int C) {
  int c = blockIdx.x * blockDim.x + threadIdx.x;
  if (c >= C) return;
  float s = g[c] * rsqrtf(v[c] + 1e-3f);
  scale[c] = s;
  shift[c] = b[c] - m[c] * s + (bias ? bias[c] * s : 0.0f);
}

// Re-pack OIDHW f32 weights into per-tap WMMA B fragments (f16), lane-contiguous:
// wp[((tap*KB + kb)*NT + nt)*32 + lane][j], j=0..15
// B(32x16) lane map: lanes 0-15 hold column n=lane, K={0..7,16..23}; lanes 16-31 hold
// column n=lane-16, K={8..15,24..31} (ISA 7.12.2 16-bit B layout). K >= CIN zero-padded.
__global__ void k_pack_w(const float* __restrict__ w, _Float16* __restrict__ wp,
                         int CIN, int COUT, int taps, int total) {
  int t = blockIdx.x * blockDim.x + threadIdx.x;
  if (t >= total) return;
  int j    = t & 15;
  int lane = (t >> 4) & 31;
  int rest = t >> 9;
  int NT = COUT >> 4;
  int nt = rest % NT; rest /= NT;
  int KB = (CIN + 31) >> 5;
  int kb  = rest % KB;
  int tap = rest / KB;
  int K = kb * 32 + ((lane >= 16) ? 8 : 0) + ((j < 8) ? j : (j + 8));
  int o = nt * 16 + (lane & 15);
  float val = 0.0f;
  if (K < CIN) val = w[(o * CIN + K) * taps + tap];   // OIDHW, spatial flattened = tap
  wp[t] = (_Float16)val;
}

// Build level-0 index grid + f16 copy of the 4-channel input features.
__global__ void k_scatter_input(const float* __restrict__ vf, const int* __restrict__ vc,
                                int* __restrict__ grid, _Float16* __restrict__ f0,
                                int N, int H, int W) {
  int i = blockIdx.x * blockDim.x + threadIdx.x;
  if (i >= N) return;
  int z = vc[i*4+1], y = vc[i*4+2], x = vc[i*4+3];
  grid[(z*H + y)*W + x] = i;
  float4 t = *(const float4*)(vf + (size_t)i * 4);
  union { _Float16 h[4]; uint2 q; } u;
  u.h[0] = (_Float16)t.x; u.h[1] = (_Float16)t.y;
  u.h[2] = (_Float16)t.z; u.h[3] = (_Float16)t.w;
  *(uint2*)(f0 + (size_t)i * 4) = u.q;
}

// Active-site dilation for strided SparseConv: output cell active iff any active
// input in its receptive field. Appends compacted coord list via atomic counter
// and writes the output-level index grid (pre-filled with -1).
__global__ void k_dilate(const int* __restrict__ gin, int iD, int iH, int iW,
                         int* __restrict__ gout, int* __restrict__ coords,
                         int* __restrict__ counter,
                         int oD, int oH, int oW, int KD, int KH, int KW,
                         int sz, int sy, int sx, int pz, int py, int px) {
  int t = blockIdx.x * blockDim.x + threadIdx.x;
  int total = oD * oH * oW;
  if (t >= total) return;
  int x = t % oW;
  int y = (t / oW) % oH;
  int z = t / (oW * oH);
  bool act = false;
  for (int kd = 0; kd < KD; ++kd) {
    int iz = z * sz - pz + kd;
    if ((unsigned)iz >= (unsigned)iD) continue;
    for (int kh = 0; kh < KH; ++kh) {
      int iy = y * sy - py + kh;
      if ((unsigned)iy >= (unsigned)iH) continue;
      for (int kw = 0; kw < KW; ++kw) {
        int ix = x * sx - px + kw;
        if ((unsigned)ix >= (unsigned)iW) continue;
        if (gin[(iz*iH + iy)*iW + ix] >= 0) act = true;
      }
    }
  }
  if (act) {
    int id = atomicAdd(counter, 1);
    gout[t] = id;
    coords[id*4+0] = 0;
    coords[id*4+1] = z;
    coords[id*4+2] = y;
    coords[id*4+3] = x;
  }
}

// ---------------------------------------------------------------------------
// Gather-GEMM sparse conv: one wave32 computes a 16(vox) x 16(chan) tile.
// A (16x32 f16) lane map (ISA 7.12.2): lane<16 holds row M=lane,
//   K = {kb*32+0..7, kb*32+16..23}; lane>=16 holds row M=lane-16,
//   K = {kb*32+8..15, kb*32+24..31}.
// Both K runs are contiguous 8-half (16B) spans of the feature row, so the
// gather is exactly two global_load_b128 per lane per WMMA (b128/b64 + zero
// padding for the CIN=16 / CIN=4 cases, resolved at compile time).
// Epilogue: fused BN scale/shift, optional residual, ReLU; f16 feature store or
// f32 dense NCDHW scatter (final layer).
// ---------------------------------------------------------------------------
template<int CIN, int COUT, int KD, int KH, int KW>
__global__ void __launch_bounds__((COUT / 16) * 32)
k_sparse_conv_wmma(const _Float16* __restrict__ fin,
                   const int* __restrict__ ocoords,
                   const int* __restrict__ ncnt, int nlit,
                   const int* __restrict__ gin, int iD, int iH, int iW,
                   int sz, int sy, int sx, int pz, int py, int px,
                   const _Float16* __restrict__ wp,
                   const float* __restrict__ scale, const float* __restrict__ shift,
                   const _Float16* __restrict__ resid,
                   _Float16* __restrict__ fout,
                   float* __restrict__ dout, int oD, int oH, int oW) {
  constexpr int KB = (CIN + 31) / 32;
  constexpr int NT = COUT / 16;
  constexpr int TAPS = KD * KH * KW;

  const int N = ncnt ? *ncnt : nlit;
  if ((int)blockIdx.x * 16 >= N) return;          // uniform early-out past active set

  const int lane = threadIdx.x & 31;
  const int nt   = threadIdx.x >> 5;              // which 16-channel tile
  const int row  = lane & 15;
  const int vox  = blockIdx.x * 16 + row;         // A-matrix row owned by this lane
  const bool vr  = vox < N;
  const bool lolane = lane < 16;
  const int Klo = lolane ? 0 : 8;

  int oz = 0, oy = 0, ox = 0;
  if (vr) { oz = ocoords[vox*4+1]; oy = ocoords[vox*4+2]; ox = ocoords[vox*4+3]; }

  v8f acc = {0.f, 0.f, 0.f, 0.f, 0.f, 0.f, 0.f, 0.f};

  for (int tap = 0; tap < TAPS; ++tap) {
    const int kd = tap / (KH * KW);
    const int rr = tap % (KH * KW);
    const int kh = rr / KW, kw = rr % KW;
    const int iz = oz * sz - pz + kd;
    const int iy = oy * sy - py + kh;
    const int ix = ox * sx - px + kw;

    int mI = -1;
    if (vr && (unsigned)iz < (unsigned)iD && (unsigned)iy < (unsigned)iH &&
        (unsigned)ix < (unsigned)iW)
      mI = gin[(iz * iH + iy) * iW + ix];

    const _Float16* fr = fin + (size_t)(mI >= 0 ? mI : 0) * CIN;

    if (tap + 1 < TAPS)                           // next tap's B fragments -> global_prefetch
      __builtin_prefetch(wp + (size_t)(tap + 1) * KB * NT * 512, 0, 0);

#pragma unroll
    for (int kb = 0; kb < KB; ++kb) {
      uint4 lo{}, hi{};
      if constexpr (CIN >= 32) {
        if (mI >= 0) {
          lo = *(const uint4*)(fr + kb * 32 + Klo);        // K = kb*32+Klo .. +7
          hi = *(const uint4*)(fr + kb * 32 + Klo + 16);   // K = kb*32+Klo+16 .. +23
        }
      } else if constexpr (CIN == 16) {
        if (mI >= 0) lo = *(const uint4*)(fr + Klo);       // hi: K>=16 -> zero pad
      } else {  // CIN == 4
        if (mI >= 0 && lolane) {
          uint2 t = *(const uint2*)fr;                     // K=0..3, rest zero pad
          lo.x = t.x; lo.y = t.y;
        }
      }
      union { v16h v; uint4 q[2]; } af;
      af.q[0] = lo; af.q[1] = hi;

      const v16h bfrag =
          *(const v16h*)(wp + (size_t)(((tap * KB + kb) * NT + nt) * 32 + lane) * 16);
      acc = __builtin_amdgcn_wmma_f32_16x16x32_f16(false, af.v, false, bfrag,
                                                   (short)0, acc, false, false);
    }
  }

  // D layout: VGPR r -> (M = r | r+8 by half-wave, N = lane&15)
  const int ch = nt * 16 + (lane & 15);
  const float sc = scale[ch], sh = shift[ch];
#pragma unroll
  for (int r = 0; r < 8; ++r) {
    const int mrow = lolane ? r : (r + 8);
    const int ov = blockIdx.x * 16 + mrow;
    if (ov < N) {
      float val = acc[r] * sc + sh;
      if (resid) val += (float)resid[(size_t)ov * COUT + ch];
      val = val > 0.f ? val : 0.f;                // ReLU (every conv in this net)
      if (fout) {
        fout[(size_t)ov * COUT + ch] = (_Float16)val;
      } else {
        const int z2 = ocoords[ov*4+1], y2 = ocoords[ov*4+2], x2 = ocoords[ov*4+3];
        dout[((size_t)ch * oD + z2) * (size_t)(oH * oW) + (size_t)y2 * oW + x2] = val;
      }
    }
  }
}

// ---------------------------------------------------------------------------
// host orchestration
// ---------------------------------------------------------------------------
extern "C" void kernel_launch(void* const* d_in, const int* in_sizes, int n_in,
                              void* d_out, int out_size, void* d_ws, size_t ws_size,
                              hipStream_t stream) {
  (void)n_in; (void)ws_size;

  const int D0 = 41, H0 = 216, W0 = 216;
  const int D1 = 21, H1 = 108, W1 = 108;
  const int D2 = 11, H2 = 54,  W2 = 54;
  const int D3 = 5,  H3 = 27,  W3 = 27;
  const int D4 = 2,  H4 = 27,  W4 = 27;
  const int CAP0 = in_sizes[0] / 4;                 // 30000 voxels, C_IN=4
  const int CAP1 = D1*H1*W1, CAP2 = D2*H2*W2, CAP3 = D3*H3*W3, CAP4 = D4*H4*W4;

  const float* vfeat  = (const float*)d_in[0];
  const int*   vcoord = (const int*)d_in[1];

  // ---- parse params (insertion-order DFS flatten of the nested dict) ----
  int cur = 2;
  auto NF = [&]() { return (const float*)d_in[cur++]; };

  struct Conv {
    const float *w, *bias, *g, *b, *m, *v;
    int cin, cout, taps;
    _Float16* wp; float *sc, *sh;
  };
  Conv cv[21]; int ncv = 0;
  auto addconv = [&](const float* w, const float* bias, const float* g, const float* b,
                     const float* m, const float* v, int cin, int cout, int taps) {
    Conv c; c.w = w; c.bias = bias; c.g = g; c.b = b; c.m = m; c.v = v;
    c.cin = cin; c.cout = cout; c.taps = taps;
    c.wp = nullptr; c.sc = nullptr; c.sh = nullptr;
    cv[ncv++] = c;
  };
  auto addblk = [&](int c) {
    const float* w1 = NF(); const float* b1 = NF();
    const float* g1 = NF(); const float* bb1 = NF(); const float* m1 = NF(); const float* v1 = NF();
    const float* w2 = NF(); const float* b2 = NF();
    const float* g2 = NF(); const float* bb2 = NF(); const float* m2 = NF(); const float* v2 = NF();
    addconv(w1, b1, g1, bb1, m1, v1, c, c, 27);
    addconv(w2, b2, g2, bb2, m2, v2, c, c, 27);
  };
  { // conv_input (4->16, no bias)
    const float* w = NF();
    const float* g = NF(); const float* b = NF(); const float* m = NF(); const float* v = NF();
    addconv(w, nullptr, g, b, m, v, 4, 16, 27);
  }
  addblk(16); addblk(16);                           // conv1
  auto addstage = [&](int cin, int cout) {          // down conv + 2 blocks
    const float* dw = NF();
    const float* g = NF(); const float* b = NF(); const float* m = NF(); const float* v = NF();
    addconv(dw, nullptr, g, b, m, v, cin, cout, 27);
    addblk(cout); addblk(cout);
  };
  addstage(16, 32);                                 // conv2  cv[5..9]
  addstage(32, 64);                                 // conv3  cv[10..14]
  addstage(64, 64);                                 // conv4  cv[15..19]
  { // conv_out (64->128, k(3,1,1), no bias)        // cv[20]
    const float* w = NF();
    const float* g = NF(); const float* b = NF(); const float* m = NF(); const float* v = NF();
    addconv(w, nullptr, g, b, m, v, 64, 128, 3);
  }

  // ---- workspace layout (bump allocator, deterministic per call) ----
  size_t off = 0; char* base = (char*)d_ws;
  auto walloc = [&](size_t bytes) -> void* {
    off = (off + 255) & ~(size_t)255;
    void* p = base + off; off += bytes; return p;
  };
  int* grid0 = (int*)walloc((size_t)D0*H0*W0*4);
  int* grid1 = (int*)walloc((size_t)CAP1*4);
  int* grid2 = (int*)walloc((size_t)CAP2*4);
  int* grid3 = (int*)walloc((size_t)CAP3*4);
  int* grid4 = (int*)walloc((size_t)CAP4*4);
  int* coords1 = (int*)walloc((size_t)CAP1*16);
  int* coords2 = (int*)walloc((size_t)CAP2*16);
  int* coords3 = (int*)walloc((size_t)CAP3*16);
  int* coords4 = (int*)walloc((size_t)CAP4*16);
  int* counters = (int*)walloc(16);
  _Float16* fin0 = (_Float16*)walloc((size_t)CAP0*4*2);
  _Float16 *f0[3], *f1[3], *f2[3], *f3[3];
  for (int i = 0; i < 3; ++i) f0[i] = (_Float16*)walloc((size_t)CAP0*16*2);
  for (int i = 0; i < 3; ++i) f1[i] = (_Float16*)walloc((size_t)CAP1*32*2);
  for (int i = 0; i < 3; ++i) f2[i] = (_Float16*)walloc((size_t)CAP2*64*2);
  for (int i = 0; i < 3; ++i) f3[i] = (_Float16*)walloc((size_t)CAP3*64*2);
  for (int i = 0; i < ncv; ++i) {
    int KB = (cv[i].cin + 31) / 32, NT = cv[i].cout / 16;
    cv[i].wp = (_Float16*)walloc((size_t)cv[i].taps * KB * NT * 512 * 2);
    cv[i].sc = (float*)walloc((size_t)cv[i].cout * 4);
    cv[i].sh = (float*)walloc((size_t)cv[i].cout * 4);
  }

  // ---- per-call init (graph-capture safe: kernels only) ----
  auto fillI = [&](int* p, int v, int n) {
    k_fill_i32<<<(n + 255) / 256, 256, 0, stream>>>(p, v, n);
  };
  fillI(grid0, -1, D0*H0*W0);
  fillI(grid1, -1, CAP1);
  fillI(grid2, -1, CAP2);
  fillI(grid3, -1, CAP3);
  fillI(grid4, -1, CAP4);
  fillI(counters, 0, 4);
  k_fill_f32<<<(out_size + 255) / 256, 256, 0, stream>>>((float*)d_out, 0.f, out_size);

  // fold BN + pack weights for all 21 convs
  for (int i = 0; i < ncv; ++i) {
    k_fold_bn<<<(cv[i].cout + 63) / 64, 64, 0, stream>>>(
        cv[i].g, cv[i].b, cv[i].m, cv[i].v, cv[i].bias, cv[i].sc, cv[i].sh, cv[i].cout);
    int KB = (cv[i].cin + 31) / 32, NT = cv[i].cout / 16;
    int total = cv[i].taps * KB * NT * 512;
    k_pack_w<<<(total + 255) / 256, 256, 0, stream>>>(
        cv[i].w, cv[i].wp, cv[i].cin, cv[i].cout, cv[i].taps, total);
  }

  // level-0 index grid + f16 input features
  k_scatter_input<<<(CAP0 + 255) / 256, 256, 0, stream>>>(vfeat, vcoord, grid0, fin0,
                                                          CAP0, H0, W0);

  // active-set dilation for the strided convs
  k_dilate<<<(CAP1 + 255) / 256, 256, 0, stream>>>(grid0, D0, H0, W0, grid1, coords1,
      counters + 0, D1, H1, W1, 3, 3, 3, 2, 2, 2, 1, 1, 1);
  k_dilate<<<(CAP2 + 255) / 256, 256, 0, stream>>>(grid1, D1, H1, W1, grid2, coords2,
      counters + 1, D2, H2, W2, 3, 3, 3, 2, 2, 2, 1, 1, 1);
  k_dilate<<<(CAP3 + 255) / 256, 256, 0, stream>>>(grid2, D2, H2, W2, grid3, coords3,
      counters + 2, D3, H3, W3, 3, 3, 3, 2, 2, 2, 0, 1, 1);
  k_dilate<<<(CAP4 + 255) / 256, 256, 0, stream>>>(grid3, D3, H3, W3, grid4, coords4,
      counters + 3, D4, H4, W4, 3, 1, 1, 2, 1, 1, 0, 0, 0);

  const int nb0 = (CAP0 + 15) / 16, nb1 = (CAP1 + 15) / 16, nb2 = (CAP2 + 15) / 16;
  const int nb3 = (CAP3 + 15) / 16, nb4 = (CAP4 + 15) / 16;

  // conv_input: 4->16 SubM
  k_sparse_conv_wmma<4,16,3,3,3><<<nb0, 32, 0, stream>>>(fin0, vcoord, nullptr, CAP0,
      grid0, D0,H0,W0, 1,1,1, 1,1,1, cv[0].wp, cv[0].sc, cv[0].sh,
      nullptr, f0[0], nullptr, 0,0,0);
  // conv1: two SparseBasicBlocks @16
  k_sparse_conv_wmma<16,16,3,3,3><<<nb0, 32, 0, stream>>>(f0[0], vcoord, nullptr, CAP0,
      grid0, D0,H0,W0, 1,1,1, 1,1,1, cv[1].wp, cv[1].sc, cv[1].sh,
      nullptr, f0[1], nullptr, 0,0,0);
  k_sparse_conv_wmma<16,16,3,3,3><<<nb0, 32, 0, stream>>>(f0[1], vcoord, nullptr, CAP0,
      grid0, D0,H0,W0, 1,1,1, 1,1,1, cv[2].wp, cv[2].sc, cv[2].sh,
      f0[0], f0[2], nullptr, 0,0,0);
  k_sparse_conv_wmma<16,16,3,3,3><<<nb0, 32, 0, stream>>>(f0[2], vcoord, nullptr, CAP0,
      grid0, D0,H0,W0, 1,1,1, 1,1,1, cv[3].wp, cv[3].sc, cv[3].sh,
      nullptr, f0[0], nullptr, 0,0,0);
  k_sparse_conv_wmma<16,16,3,3,3><<<nb0, 32, 0, stream>>>(f0[0], vcoord, nullptr, CAP0,
      grid0, D0,H0,W0, 1,1,1, 1,1,1, cv[4].wp, cv[4].sc, cv[4].sh,
      f0[2], f0[1], nullptr, 0,0,0);

  // conv2: 16->32 stride-2 down + two blocks @32
  k_sparse_conv_wmma<16,32,3,3,3><<<nb1, 64, 0, stream>>>(f0[1], coords1, counters+0, 0,
      grid0, D0,H0,W0, 2,2,2, 1,1,1, cv[5].wp, cv[5].sc, cv[5].sh,
      nullptr, f1[0], nullptr, 0,0,0);
  k_sparse_conv_wmma<32,32,3,3,3><<<nb1, 64, 0, stream>>>(f1[0], coords1, counters+0, 0,
      grid1, D1,H1,W1, 1,1,1, 1,1,1, cv[6].wp, cv[6].sc, cv[6].sh,
      nullptr, f1[1], nullptr, 0,0,0);
  k_sparse_conv_wmma<32,32,3,3,3><<<nb1, 64, 0, stream>>>(f1[1], coords1, counters+0, 0,
      grid1, D1,H1,W1, 1,1,1, 1,1,1, cv[7].wp, cv[7].sc, cv[7].sh,
      f1[0], f1[2], nullptr, 0,0,0);
  k_sparse_conv_wmma<32,32,3,3,3><<<nb1, 64, 0, stream>>>(f1[2], coords1, counters+0, 0,
      grid1, D1,H1,W1, 1,1,1, 1,1,1, cv[8].wp, cv[8].sc, cv[8].sh,
      nullptr, f1[0], nullptr, 0,0,0);
  k_sparse_conv_wmma<32,32,3,3,3><<<nb1, 64, 0, stream>>>(f1[0], coords1, counters+0, 0,
      grid1, D1,H1,W1, 1,1,1, 1,1,1, cv[9].wp, cv[9].sc, cv[9].sh,
      f1[2], f1[1], nullptr, 0,0,0);

  // conv3: 32->64 stride-2 down + two blocks @64
  k_sparse_conv_wmma<32,64,3,3,3><<<nb2, 128, 0, stream>>>(f1[1], coords2, counters+1, 0,
      grid1, D1,H1,W1, 2,2,2, 1,1,1, cv[10].wp, cv[10].sc, cv[10].sh,
      nullptr, f2[0], nullptr, 0,0,0);
  k_sparse_conv_wmma<64,64,3,3,3><<<nb2, 128, 0, stream>>>(f2[0], coords2, counters+1, 0,
      grid2, D2,H2,W2, 1,1,1, 1,1,1, cv[11].wp, cv[11].sc, cv[11].sh,
      nullptr, f2[1], nullptr, 0,0,0);
  k_sparse_conv_wmma<64,64,3,3,3><<<nb2, 128, 0, stream>>>(f2[1], coords2, counters+1, 0,
      grid2, D2,H2,W2, 1,1,1, 1,1,1, cv[12].wp, cv[12].sc, cv[12].sh,
      f2[0], f2[2], nullptr, 0,0,0);
  k_sparse_conv_wmma<64,64,3,3,3><<<nb2, 128, 0, stream>>>(f2[2], coords2, counters+1, 0,
      grid2, D2,H2,W2, 1,1,1, 1,1,1, cv[13].wp, cv[13].sc, cv[13].sh,
      nullptr, f2[0], nullptr, 0,0,0);
  k_sparse_conv_wmma<64,64,3,3,3><<<nb2, 128, 0, stream>>>(f2[0], coords2, counters+1, 0,
      grid2, D2,H2,W2, 1,1,1, 1,1,1, cv[14].wp, cv[14].sc, cv[14].sh,
      f2[2], f2[1], nullptr, 0,0,0);

  // conv4: 64->64 stride-2 down (pad 0,1,1) + two blocks @64
  k_sparse_conv_wmma<64,64,3,3,3><<<nb3, 128, 0, stream>>>(f2[1], coords3, counters+2, 0,
      grid2, D2,H2,W2, 2,2,2, 0,1,1, cv[15].wp, cv[15].sc, cv[15].sh,
      nullptr, f3[0], nullptr, 0,0,0);
  k_sparse_conv_wmma<64,64,3,3,3><<<nb3, 128, 0, stream>>>(f3[0], coords3, counters+2, 0,
      grid3, D3,H3,W3, 1,1,1, 1,1,1, cv[16].wp, cv[16].sc, cv[16].sh,
      nullptr, f3[1], nullptr, 0,0,0);
  k_sparse_conv_wmma<64,64,3,3,3><<<nb3, 128, 0, stream>>>(f3[1], coords3, counters+2, 0,
      grid3, D3,H3,W3, 1,1,1, 1,1,1, cv[17].wp, cv[17].sc, cv[17].sh,
      f3[0], f3[2], nullptr, 0,0,0);
  k_sparse_conv_wmma<64,64,3,3,3><<<nb3, 128, 0, stream>>>(f3[2], coords3, counters+2, 0,
      grid3, D3,H3,W3, 1,1,1, 1,1,1, cv[18].wp, cv[18].sc, cv[18].sh,
      nullptr, f3[0], nullptr, 0,0,0);
  k_sparse_conv_wmma<64,64,3,3,3><<<nb3, 128, 0, stream>>>(f3[0], coords3, counters+2, 0,
      grid3, D3,H3,W3, 1,1,1, 1,1,1, cv[19].wp, cv[19].sc, cv[19].sh,
      f3[2], f3[1], nullptr, 0,0,0);

  // conv_out: 64->128, k(3,1,1), s(2,1,1), p0 -> dense f32 [128,2,27,27]
  k_sparse_conv_wmma<64,128,3,1,1><<<nb4, 256, 0, stream>>>(f3[1], coords4, counters+3, 0,
      grid3, D3,H3,W3, 2,1,1, 0,0,0, cv[20].wp, cv[20].sc, cv[20].sh,
      nullptr, nullptr, (float*)d_out, D4, H4, W4);
}
